// LogSigRNNGenerator_23459111371238
// MI455X (gfx1250) — matrix operand order repaired and checked by reading, not compiled, not run on hardware
//
#include <hip/hip_runtime.h>
#include <hip/hip_bf16.h>

typedef _Float16 half_t;
typedef __attribute__((ext_vector_type(16))) _Float16 v16h;
typedef __attribute__((ext_vector_type(8)))  _Float16 v8h;
typedef __attribute__((ext_vector_type(8)))  float    v8f;

#define BATCH     2048
#define HIDDEN    512
#define LEN_NOISE 1000
#define WINDOW    64
#define NWIN      63
#define LSCH      6
#define KAUG      544      // 512 + 6 padded up to multiple of 32
#define MAXSLOTS  128

// ---------------------------------------------------------------------------
// Setup: build time->slot table for snapshot capture, and recompute the
// update mask from ind_low (derives u_list exactly like the reference;
// comparison margins are >= 1.4e-4 so FP-safe).
// ---------------------------------------------------------------------------
__global__ void k_setup(const int* __restrict__ ind_low,
                        const int* __restrict__ ind_max,
                        int* __restrict__ slotOf,
                        int* __restrict__ slotMax,
                        int* __restrict__ slotLow,
                        int* __restrict__ upd) {
  if (threadIdx.x != 0 || blockIdx.x != 0) return;
  for (int i = 0; i < LEN_NOISE; ++i) slotOf[i] = -1;
  int ns = 0;
  for (int k = 0; k < NWIN; ++k) {
    int im = ind_max[k];
    if (slotOf[im] < 0) slotOf[im] = ns++;
    slotMax[k] = slotOf[im];
    int il = ind_low[k];
    if (slotOf[il] < 0) slotOf[il] = ns++;
    slotLow[k] = slotOf[il];
  }
  // distinct u values (ind_low is non-decreasing) then final t = 1.0
  double ulist[NWIN + 1];
  int nu = 0;
  for (int k = 0; k < NWIN; ++k)
    if (k == 0 || ind_low[k] != ind_low[k - 1])
      ulist[nu++] = (double)ind_low[k] / 999.0;
  ulist[nu++] = 1.0;
  int ptr = 0;
  for (int i = 0; i < WINDOW; ++i) {
    double t = (double)i / 63.0;
    int u = (ptr < nu && t >= ulist[ptr]) ? 1 : 0;
    if (u) ptr++;
    upd[i] = u;
  }
}

// ---------------------------------------------------------------------------
// Log-signature recurrence: per batch element track path (3) and the
// antisymmetric cross-integral D_ij = C_ij - C_ji (3), snapshot at slots.
// ---------------------------------------------------------------------------
__global__ __launch_bounds__(256) void k_features(const float* __restrict__ z,
                                                  const int* __restrict__ slotOf,
                                                  float* __restrict__ snaps) {
  __shared__ int s_slot[LEN_NOISE];
  for (int i = threadIdx.x; i < LEN_NOISE; i += 256) s_slot[i] = slotOf[i];
  __syncthreads();
  int b = blockIdx.x * 256 + threadIdx.x;
  if (b >= BATCH) return;
  const float s = sqrtf(1.0f / 999.0f);
  float p0 = 0.f, p1 = 0.f, p2 = 0.f, d01 = 0.f, d02 = 0.f, d12 = 0.f;
  {
    int sl = s_slot[0];
    if (sl >= 0) {
      float* o = snaps + ((size_t)b * MAXSLOTS + sl) * 6;
      o[0] = 0.f; o[1] = 0.f; o[2] = 0.f; o[3] = 0.f; o[4] = 0.f; o[5] = 0.f;
    }
  }
  const float* zp = z + (size_t)b * LEN_NOISE * 3;
  for (int m = 1; m < LEN_NOISE; ++m) {
    float z0 = zp[m * 3 + 0] * s;
    float z1 = zp[m * 3 + 1] * s;
    float z2 = zp[m * 3 + 2] * s;
    d01 += p0 * z1 - p1 * z0;       // cross uses OLD path
    d02 += p0 * z2 - p2 * z0;
    d12 += p1 * z2 - p2 * z1;
    p0 += z0; p1 += z1; p2 += z2;
    int sl = s_slot[m];
    if (sl >= 0) {
      float* o = snaps + ((size_t)b * MAXSLOTS + sl) * 6;
      o[0] = p0; o[1] = p1; o[2] = p2; o[3] = d01; o[4] = d02; o[5] = d12;
    }
  }
}

// logsig[b][k][0..5] in f16 (row 0 = zeros)
__global__ __launch_bounds__(256) void k_logsig(const float* __restrict__ snaps,
                                                const int* __restrict__ slotMax,
                                                const int* __restrict__ slotLow,
                                                half_t* __restrict__ ls16) {
  int idx = blockIdx.x * 256 + threadIdx.x;
  if (idx >= BATCH * WINDOW) return;
  int b = idx / WINDOW, k = idx % WINDOW;
  half_t* o = ls16 + ((size_t)b * WINDOW + k) * LSCH;
  if (k == 0) {
    for (int c = 0; c < LSCH; ++c) o[c] = (half_t)0.f;
    return;
  }
  int kk = k - 1;
  const float* mx = snaps + ((size_t)b * MAXSLOTS + slotMax[kk]) * 6;
  const float* lw = snaps + ((size_t)b * MAXSLOTS + slotLow[kk]) * 6;
  o[0] = (half_t)mx[0]; o[1] = (half_t)mx[1]; o[2] = (half_t)mx[2];
  o[3] = (half_t)(0.5f * (mx[3] - lw[3]));
  o[4] = (half_t)(0.5f * (mx[4] - lw[4]));
  o[5] = (half_t)(0.5f * (mx[5] - lw[5]));
}

// ---------------------------------------------------------------------------
// Weight conversion: f32 [K][N] -> f16 transposed [N][K] (K padded w/ zeros)
// ---------------------------------------------------------------------------
__global__ void k_convW1(const float* __restrict__ W1, half_t* __restrict__ W1t) {
  int idx = blockIdx.x * 256 + threadIdx.x;
  if (idx >= HIDDEN * KAUG) return;
  int n = idx / KAUG, k = idx % KAUG;
  W1t[idx] = (k < HIDDEN + LSCH) ? (half_t)W1[k * HIDDEN + n] : (half_t)0.f;
}
__global__ void k_convW(const float* __restrict__ W, half_t* __restrict__ Wt) {
  int idx = blockIdx.x * 256 + threadIdx.x;
  if (idx >= HIDDEN * HIDDEN) return;
  int n = idx / HIDDEN, k = idx % HIDDEN;
  Wt[idx] = (half_t)W[k * HIDDEN + n];
}
__global__ void k_zero_h16(half_t* __restrict__ p, int n) {
  int i = blockIdx.x * 256 + threadIdx.x;
  if (i < n) p[i] = (half_t)0.f;
}

// copy step-t logsig channels into the augmented GEMM1 input
__global__ void k_prep(const half_t* __restrict__ ls16, half_t* __restrict__ Xaug, int t) {
  int b = blockIdx.x * 256 + threadIdx.x;
  if (b >= BATCH) return;
  const half_t* src = ls16 + ((size_t)b * WINDOW + t) * LSCH;
  half_t* dst = Xaug + (size_t)b * KAUG + HIDDEN;
  for (int c = 0; c < LSCH; ++c) dst[c] = src[c];
}

// ---------------------------------------------------------------------------
// WMMA GEMM: Y[M=2048][N=512] = A[M][K] * Bt[N][K]^T, f16 in / f32 acc.
// Wave tile: 16(M) x 64(N): 1 A-frag reused across 4 B-frags.
// Block mapping: all 8 waves of a block share the SAME N-group (identical
// B addresses -> WGP-cache hits), stacked along M.
// mode 0: bias+PReLU   mode 1: bias+tanh (+ optional conditional commit to Y2)
// ---------------------------------------------------------------------------
__global__ __launch_bounds__(256) void k_gemm(const half_t* __restrict__ A, int lda,
                                              const half_t* __restrict__ Bt, int ldb, int K,
                                              const float* __restrict__ bias,
                                              const float* __restrict__ alpha, int mode,
                                              half_t* __restrict__ Y,
                                              half_t* __restrict__ Y2, int ldy2,
                                              const int* __restrict__ updFlag) {
  const int lane    = threadIdx.x & 31;
  const int wave    = threadIdx.x >> 5;
  const int ngrp    = blockIdx.x & 7;          // 8 N-groups of 64 cols
  const int mblk    = blockIdx.x >> 3;         // 16 M-blocks of 128 rows
  const int m0      = (mblk * 8 + wave) * 16;  // 128 M-tiles of 16 rows
  const int n0      = ngrp * 64;
  const int halfSel = lane >> 4;   // 0: lanes 0-15, 1: lanes 16-31
  const int lid     = lane & 15;

  v8f acc0 = {}, acc1 = {}, acc2 = {}, acc3 = {};

  // A fragment: row m0+lid, per lane-half: K-chunks [off..off+7] and [off+16..off+23]
  const half_t* Abase = A + (size_t)(m0 + lid) * lda + (halfSel << 3);
  // B fragment: column n, per lane-half: K-chunks [off..off+7] and [off+16..off+23]
  const half_t* Bb0 = Bt + (size_t)(n0 +  0 + lid) * ldb + (halfSel << 3);
  const half_t* Bb1 = Bt + (size_t)(n0 + 16 + lid) * ldb + (halfSel << 3);
  const half_t* Bb2 = Bt + (size_t)(n0 + 32 + lid) * ldb + (halfSel << 3);
  const half_t* Bb3 = Bt + (size_t)(n0 + 48 + lid) * ldb + (halfSel << 3);

  for (int kk = 0; kk < K; kk += 32) {
    v8h alo = *(const v8h*)(Abase + kk);
    v8h ahi = *(const v8h*)(Abase + kk + 16);
    v16h a = __builtin_shufflevector(alo, ahi, 0, 1, 2, 3, 4, 5, 6, 7,
                                     8, 9, 10, 11, 12, 13, 14, 15);
    v16h b0 = __builtin_shufflevector(*(const v8h*)(Bb0 + kk), *(const v8h*)(Bb0 + kk + 16), 0, 1, 2, 3, 4, 5, 6, 7, 8, 9, 10, 11, 12, 13, 14, 15);
    v16h b1 = __builtin_shufflevector(*(const v8h*)(Bb1 + kk), *(const v8h*)(Bb1 + kk + 16), 0, 1, 2, 3, 4, 5, 6, 7, 8, 9, 10, 11, 12, 13, 14, 15);
    v16h b2 = __builtin_shufflevector(*(const v8h*)(Bb2 + kk), *(const v8h*)(Bb2 + kk + 16), 0, 1, 2, 3, 4, 5, 6, 7, 8, 9, 10, 11, 12, 13, 14, 15);
    v16h b3 = __builtin_shufflevector(*(const v8h*)(Bb3 + kk), *(const v8h*)(Bb3 + kk + 16), 0, 1, 2, 3, 4, 5, 6, 7, 8, 9, 10, 11, 12, 13, 14, 15);
    acc0 = __builtin_amdgcn_wmma_f32_16x16x32_f16(false, a, false, b0, (short)0, acc0, false, false);
    acc1 = __builtin_amdgcn_wmma_f32_16x16x32_f16(false, a, false, b1, (short)0, acc1, false, false);
    acc2 = __builtin_amdgcn_wmma_f32_16x16x32_f16(false, a, false, b2, (short)0, acc2, false, false);
    acc3 = __builtin_amdgcn_wmma_f32_16x16x32_f16(false, a, false, b3, (short)0, acc3, false, false);
  }

  const int   updv = (Y2 != nullptr && updFlag != nullptr) ? updFlag[0] : 0;
  const float al   = (mode == 0) ? alpha[0] : 0.f;

#define EPILOGUE(ACC, J)                                                        \
  do {                                                                          \
    int col = n0 + (J) * 16 + lid;                                              \
    float bv = bias[col];                                                       \
    _Pragma("unroll")                                                           \
    for (int i = 0; i < 8; ++i) {                                               \
      int row = m0 + i + (halfSel << 3);                                        \
      float v = (ACC)[i] + bv;                                                  \
      v = (mode == 0) ? ((v >= 0.f) ? v : al * v) : tanhf(v);                   \
      half_t hv = (half_t)v;                                                    \
      Y[(size_t)row * HIDDEN + col] = hv;                                       \
      if (updv) Y2[(size_t)row * ldy2 + col] = hv;                              \
    }                                                                           \
  } while (0)

  EPILOGUE(acc0, 0);
  EPILOGUE(acc1, 1);
  EPILOGUE(acc2, 2);
  EPILOGUE(acc3, 3);
#undef EPILOGUE
}

// out[b][t][0..2] = h[b] @ Wout : one wave per batch row, coalesced loads,
// wave32 shuffle reduction.
__global__ __launch_bounds__(256) void k_out(const half_t* __restrict__ H,
                                             const float* __restrict__ Wout,
                                             float* __restrict__ out, int t) {
  const int lane = threadIdx.x & 31;
  const int wave = threadIdx.x >> 5;
  const int b    = blockIdx.x * 8 + wave;
  const half_t* h = H + (size_t)b * HIDDEN;
  float a0 = 0.f, a1 = 0.f, a2 = 0.f;
  for (int j = lane; j < HIDDEN; j += 32) {
    float hv = (float)h[j];
    a0 += hv * Wout[j * 3 + 0];
    a1 += hv * Wout[j * 3 + 1];
    a2 += hv * Wout[j * 3 + 2];
  }
  for (int o = 16; o > 0; o >>= 1) {
    a0 += __shfl_down(a0, o, 32);
    a1 += __shfl_down(a1, o, 32);
    a2 += __shfl_down(a2, o, 32);
  }
  if (lane == 0) {
    float* o = out + (size_t)b * WINDOW * 3 + t * 3;
    o[0] = a0; o[1] = a1; o[2] = a2;
  }
}

// ---------------------------------------------------------------------------
extern "C" void kernel_launch(void* const* d_in, const int* in_sizes, int n_in,
                              void* d_out, int out_size, void* d_ws, size_t ws_size,
                              hipStream_t stream) {
  (void)in_sizes; (void)n_in; (void)out_size; (void)ws_size;
  const float* z       = (const float*)d_in[0];
  const int*   ind_low = (const int*)d_in[1];
  const int*   ind_max = (const int*)d_in[2];
  /* d_in[3] = update_mask: recomputed on device (bool dtype ambiguity) */
  const float* W1   = (const float*)d_in[4];
  const float* b1   = (const float*)d_in[5];
  const float* a1   = (const float*)d_in[6];
  const float* W2   = (const float*)d_in[7];
  const float* b2   = (const float*)d_in[8];
  const float* a2   = (const float*)d_in[9];
  const float* W3   = (const float*)d_in[10];
  const float* b3   = (const float*)d_in[11];
  const float* Wout = (const float*)d_in[12];
  float* out = (float*)d_out;

  char* ws = (char*)d_ws;
  size_t off = 0;
  auto carve = [&](size_t bytes) -> char* {
    char* p = ws + off;
    off = (off + bytes + 255) & ~(size_t)255;
    return p;
  };
  int*    slotOf  = (int*)carve(LEN_NOISE * 4);
  int*    slotMax = (int*)carve(NWIN * 4);
  int*    slotLow = (int*)carve(NWIN * 4);
  int*    upd     = (int*)carve(WINDOW * 4);
  float*  snaps   = (float*)carve((size_t)BATCH * MAXSLOTS * 6 * 4);
  half_t* ls16    = (half_t*)carve((size_t)BATCH * WINDOW * LSCH * 2);
  half_t* W1t     = (half_t*)carve((size_t)HIDDEN * KAUG * 2);
  half_t* W2t     = (half_t*)carve((size_t)HIDDEN * HIDDEN * 2);
  half_t* W3t     = (half_t*)carve((size_t)HIDDEN * HIDDEN * 2);
  half_t* Xaug    = (half_t*)carve((size_t)BATCH * KAUG * 2);
  half_t* X1      = (half_t*)carve((size_t)BATCH * HIDDEN * 2);
  half_t* X2      = (half_t*)carve((size_t)BATCH * HIDDEN * 2);
  half_t* Hbuf    = (half_t*)carve((size_t)BATCH * HIDDEN * 2);

  k_setup<<<1, 1, 0, stream>>>(ind_low, ind_max, slotOf, slotMax, slotLow, upd);
  k_convW1<<<(HIDDEN * KAUG + 255) / 256, 256, 0, stream>>>(W1, W1t);
  k_convW<<<(HIDDEN * HIDDEN + 255) / 256, 256, 0, stream>>>(W2, W2t);
  k_convW<<<(HIDDEN * HIDDEN + 255) / 256, 256, 0, stream>>>(W3, W3t);
  k_zero_h16<<<(BATCH * KAUG + 255) / 256, 256, 0, stream>>>(Xaug, BATCH * KAUG);
  k_features<<<BATCH / 256, 256, 0, stream>>>(z, slotOf, snaps);
  k_logsig<<<(BATCH * WINDOW + 255) / 256, 256, 0, stream>>>(snaps, slotMax, slotLow, ls16);

  for (int t = 0; t < WINDOW; ++t) {
    k_prep<<<BATCH / 256, 256, 0, stream>>>(ls16, Xaug, t);
    k_gemm<<<128, 256, 0, stream>>>(Xaug, KAUG, W1t, KAUG, KAUG, b1, a1, 0,
                                    X1, nullptr, 0, nullptr);
    k_gemm<<<128, 256, 0, stream>>>(X1, HIDDEN, W2t, HIDDEN, HIDDEN, b2, a2, 0,
                                    X2, nullptr, 0, nullptr);
    k_gemm<<<128, 256, 0, stream>>>(X2, HIDDEN, W3t, HIDDEN, HIDDEN, b3, nullptr, 1,
                                    Hbuf, Xaug, KAUG, upd + t);
    k_out<<<BATCH / 8, 256, 0, stream>>>(Hbuf, Wout, out, t);
  }
}
